// HashNeRF_25245817765926
// MI455X (gfx1250) — matrix operand order repaired and checked by reading, not compiled, not run on hardware
//
#include <hip/hip_runtime.h>
#include <cstdint>
#include <cstddef>

// ---------------------------------------------------------------------------
// HashNeRF forward on gfx1250: WMMA f32<-f16 GEMM pipeline.
//   D = A x B + C with A = weight tile (16 outcols x 32 K, pre-swizzled,
//   staged into LDS with GLOBAL_LOAD_ASYNC_TO_LDS_B128), B = activation tile
//   (32 K x 16 rows, contiguous global loads). Each wave computes 16 rows x
//   128 cols (8 tiles, 64 acc VGPRs) for ~4 waves/SIMD so LDS latency is
//   hidden by wave interleave. Lane accumulators are 8 consecutive output
//   columns of one row -> packed b128 stores.
// ---------------------------------------------------------------------------

typedef __attribute__((ext_vector_type(16))) _Float16 v16h;
typedef __attribute__((ext_vector_type(8)))  float    v8f;

#define NPOINTS  262144       // B*P
#define RAYS     2048
#define NLVL     16
#define LOG2T    19
#define TSIZE    (1 << LOG2T)
#define NHARM    60
#define HD       360          // 6*NHARM
#define ENCKP    416          // 392 padded to 13*32
#define LATKP    640          // 616 padded to 20*32
#define WIDTH    256
#define DEPAD    384          // 360 padded

union AU  { uint4 u[2]; v16h v; };
union H4  { _Float16 h[4]; uint2 u; };
union H8  { _Float16 h[8]; uint4 u; };

// --------------------------- harmonic embedding ----------------------------
// Layout: row[d*60 + k] = sin(x_d * 2^k), row[180 + d*60 + k] = cos(...).
// Frequencies reach 2^59 where float phase is meaningless, so hardware
// v_sin/v_cos (TRANS pipe, co-executes with VALU) is the right choice.
__device__ __forceinline__ void write_harmonic_v(_Float16* __restrict__ row,
                                                 float nx, float ny, float nz) {
  const float n3[3] = {nx, ny, nz};
  #pragma unroll
  for (int d = 0; d < 3; ++d) {
    const float v = n3[d];
    float f = 1.0f;
    #pragma unroll 3
    for (int c = 0; c < 15; ++c) {          // 15 chunks of 4 frequencies
      H4 s, cc;
      #pragma unroll
      for (int j = 0; j < 4; ++j) {
        const float a = v * f;
        s.h[j]  = (_Float16)__sinf(a);
        cc.h[j] = (_Float16)__cosf(a);
        f *= 2.0f;
      }
      *(uint2*)(row + d * 60 + c * 4)       = s.u;
      *(uint2*)(row + 180 + d * 60 + c * 4) = cc.u;
    }
  }
}

// --------------------------- point encoder --------------------------------
__global__ __launch_bounds__(256) void encode_kernel(
    const float* __restrict__ origins, const float* __restrict__ dirs,
    const float* __restrict__ lengths, const float* __restrict__ tables,
    _Float16* __restrict__ ENC, int n0) {
  const int i = blockIdx.x * 256 + threadIdx.x;           // chunk-local row
  const long gi = (long)n0 + i;                           // global point id
  const int ray = (int)(gi >> 7);

  const float ox = origins[ray * 3 + 0], oy = origins[ray * 3 + 1], oz = origins[ray * 3 + 2];
  const float dx = dirs[ray * 3 + 0],    dy = dirs[ray * 3 + 1],    dz = dirs[ray * 3 + 2];
  const float t = lengths[gi];
  const float px = ox + dx * t, py = oy + dy * t, pz = oz + dz * t;

  _Float16* row = ENC + (size_t)i * ENCKP;

  // ---- hash-grid features (16 levels x 2) into registers, then b128 stores
  const float cxp = fminf(fmaxf(px, -1.0f), 5.0f);
  const float cyp = fminf(fmaxf(py, -1.0f), 5.0f);
  const float czp = fminf(fmaxf(pz, -1.0f), 5.0f);
  const float2* tab = (const float2*)tables;

  union { _Float16 h[32]; uint4 u[4]; } hf;
  #pragma unroll
  for (int l = 0; l < NLVL; ++l) {
    const float res  = floorf(16.0f * exp2f((float)l * (1.0f / 3.0f)));  // 16 * 2^(l/3)
    const float grid = 6.0f / res;
    const int blx = (int)floorf((cxp + 1.0f) / grid);
    const int bly = (int)floorf((cyp + 1.0f) / grid);
    const int blz = (int)floorf((czp + 1.0f) / grid);
    const float wx = (cxp - ((float)blx * grid - 1.0f)) / grid;
    const float wy = (cyp - ((float)bly * grid - 1.0f)) / grid;
    const float wz = (czp - ((float)blz * grid - 1.0f)) / grid;
    const unsigned bx = (unsigned)blx, by = (unsigned)bly, bz = (unsigned)blz;

    float f0 = 0.0f, f1 = 0.0f;
    #pragma unroll
    for (int c = 0; c < 8; ++c) {
      const unsigned ci = bx + ((c >> 2) & 1);
      const unsigned cj = by + ((c >> 1) & 1);
      const unsigned ck = bz + (c & 1);
      unsigned h = ci ^ (cj * 2654435761u) ^ (ck * 805459861u);
      h &= (unsigned)(TSIZE - 1);
      const float2 e = tab[(size_t)l * TSIZE + h];
      const float w = ((c & 4) ? wx : 1.0f - wx) *
                      ((c & 2) ? wy : 1.0f - wy) *
                      ((c & 1) ? wz : 1.0f - wz);
      f0 += e.x * w; f1 += e.y * w;
    }
    hf.h[2 * l]     = (_Float16)f0;
    hf.h[2 * l + 1] = (_Float16)f1;
  }
  #pragma unroll
  for (int c = 0; c < 4; ++c) *(uint4*)(row + c * 8) = hf.u[c];

  // ---- harmonic embedding of normalized point
  const float nrm = fmaxf(sqrtf(px * px + py * py + pz * pz), 1e-12f);
  write_harmonic_v(row + 32, px / nrm, py / nrm, pz / nrm);

  // ---- zero K-padding (cols 392..415)
  const uint4 z = {0u, 0u, 0u, 0u};
  #pragma unroll
  for (int c = 0; c < 3; ++c) *(uint4*)(row + 392 + c * 8) = z;
}

// ------------------------ per-ray direction embedding ----------------------
__global__ __launch_bounds__(256) void dirembed_kernel(
    const float* __restrict__ dirs, _Float16* __restrict__ DE) {
  const int r = blockIdx.x * 256 + threadIdx.x;
  if (r >= RAYS) return;
  const float dx = dirs[r * 3 + 0], dy = dirs[r * 3 + 1], dz = dirs[r * 3 + 2];
  const float nrm = fmaxf(sqrtf(dx * dx + dy * dy + dz * dz), 1e-12f);
  _Float16* row = DE + (size_t)r * DEPAD;
  write_harmonic_v(row, dx / nrm, dy / nrm, dz / nrm);
  const uint4 z = {0u, 0u, 0u, 0u};
  #pragma unroll
  for (int c = 0; c < 3; ++c) *(uint4*)(row + HD + c * 8) = z;
}

// ------------------------- weight swizzler ---------------------------------
// Convert f32 row-major W[K x 256] into f16 WMMA *A-fragment* order
// (weights act as the A matrix: M = output column, K = input dim):
//   layout [ktile][coltile(16)][lane(32)][16 halves]
//   k_local = idx + hi*16 ; rows with k >= K are zero-padded.
__global__ __launch_bounds__(256) void swizzle_w_kernel(
    const float* __restrict__ W, int K, _Float16* __restrict__ dst) {
  const int kt = blockIdx.x;
  #pragma unroll 4
  for (int j = 0; j < 32; ++j) {
    const int id   = j * 256 + threadIdx.x;   // 0..8191, coalesced writes
    const int ct   = id >> 9;
    const int rem  = id & 511;
    const int lane = rem >> 4;
    const int idx  = rem & 15;
    const int klocal = idx + ((lane >> 4) << 4);
    const int k = kt * 32 + klocal;
    const int m = (ct << 4) + (lane & 15);
    const float v = (k < K) ? W[(size_t)k * WIDTH + m] : 0.0f;
    dst[(size_t)kt * 8192 + id] = (_Float16)v;
  }
}

// --------------------------- WMMA GEMM layer -------------------------------
// Async DMA of a 16 KB weight K-tile into LDS: no VGPR data, tracked by
// ASYNCcnt, fully overlaps the WMMA block. IOFFSET applies to both the
// global and the LDS address (CDNA5 ISA).
__device__ __forceinline__ void stage_async16k(const _Float16* __restrict__ g,
                                               _Float16* __restrict__ s, int tid) {
  const unsigned long long ga =
      (unsigned long long)(uintptr_t)g + (unsigned)(tid * 16);
  const unsigned ls =                       // low 32 bits of flat = LDS offset
      (unsigned)(uintptr_t)s + (unsigned)(tid * 16);
  asm volatile(
      "global_load_async_to_lds_b128 %0, %1, off\n\t"
      "global_load_async_to_lds_b128 %0, %1, off offset:4096\n\t"
      "global_load_async_to_lds_b128 %0, %1, off offset:8192\n\t"
      "global_load_async_to_lds_b128 %0, %1, off offset:12288"
      :: "v"(ls), "v"(ga)
      : "memory");
}

__device__ __forceinline__ void wait_async_all() {
  asm volatile("s_wait_asynccnt 0x0" ::: "memory");
}

__device__ __forceinline__ void ld_fragA(AU& w, const _Float16* __restrict__ bb,
                                         int lane, int ct) {
  const _Float16* p = bb + ((ct * 32 + lane) << 4);   // 32B per lane
  w.u[0] = *(const uint4*)(p);
  w.u[1] = *(const uint4*)(p + 8);
}

// Y[rows x 256] = act(X[rows x ldx] @ W + bias).
// Block = 256 threads (8 waves) -> 64 rows x 256 cols. Waves: 4 row-groups
// x 2 column-groups; each wave does 16 rows x 128 cols = 8 WMMA tiles,
// 8 v_wmma_f32_16x16x32_f16 per K-step, 64 accumulator VGPRs. Weight K-tile
// (16 KB) double-buffered through LDS via async DMA, shared by all 8 waves.
__global__ __launch_bounds__(256) void gemm_wmma_256(
    const _Float16* __restrict__ X, int ldx, int ktiles,
    const _Float16* __restrict__ Wswz, const float* __restrict__ bias,
    _Float16* __restrict__ Y, int do_relu) {
  __shared__ __align__(16) _Float16 smem[2][8192];   // 2 x 16 KB K-tiles

  const int tid  = threadIdx.x;
  const int lane = tid & 31;
  const int wave = tid >> 5;
  const int colg = wave & 1;                         // column half: 0 / 1
  const int rowg = wave >> 1;                        // row group: 0..3
  const long rowBase = (long)blockIdx.x * 64 + rowg * 16;
  const int hi = lane >> 4;

  v8f acc[8];
  const v8f vzero = {0.f, 0.f, 0.f, 0.f, 0.f, 0.f, 0.f, 0.f};
  #pragma unroll
  for (int ct = 0; ct < 8; ++ct) acc[ct] = vzero;

  // Activation (B) fragment: lane holds 32 contiguous bytes of its row:
  //   row = rowBase + (lane&15), K offset = kt*32 + hi*16.
  const _Float16* bptr = X + (size_t)(rowBase + (lane & 15)) * ldx + hi * 16;

  stage_async16k(Wswz, smem[0], tid);
  AU bcur;
  bcur.u[0] = *(const uint4*)(bptr);
  bcur.u[1] = *(const uint4*)(bptr + 8);
  wait_async_all();
  __syncthreads();

  int buf = 0;
  for (int kt = 0; kt < ktiles; ++kt) {
    AU bnxt = bcur;
    if (kt + 1 < ktiles) {
      stage_async16k(Wswz + (size_t)(kt + 1) * 8192, smem[buf ^ 1], tid);
      const _Float16* np = bptr + (size_t)(kt + 1) * 32;
      bnxt.u[0] = *(const uint4*)(np);
      bnxt.u[1] = *(const uint4*)(np + 8);
    }
    if (kt + 2 < ktiles)                                   // global_prefetch_b8
      __builtin_prefetch(Wswz + (size_t)(kt + 2) * 8192 + tid * 32, 0, 0);

    // This wave's half of the tile: fragments colg*8 .. colg*8+7.
    const _Float16* bb = smem[buf] + (size_t)colg * 4096;
    #pragma unroll
    for (int g = 0; g < 2; ++g) {
      AU w0, w1, w2, w3;                 // 4 distinct in-flight fragments
      ld_fragA(w0, bb, lane, g * 4 + 0);
      ld_fragA(w1, bb, lane, g * 4 + 1);
      ld_fragA(w2, bb, lane, g * 4 + 2);
      ld_fragA(w3, bb, lane, g * 4 + 3);
      acc[g * 4 + 0] = __builtin_amdgcn_wmma_f32_16x16x32_f16(
          false, w0.v, false, bcur.v, (short)0, acc[g * 4 + 0], false, false);
      acc[g * 4 + 1] = __builtin_amdgcn_wmma_f32_16x16x32_f16(
          false, w1.v, false, bcur.v, (short)0, acc[g * 4 + 1], false, false);
      acc[g * 4 + 2] = __builtin_amdgcn_wmma_f32_16x16x32_f16(
          false, w2.v, false, bcur.v, (short)0, acc[g * 4 + 2], false, false);
      acc[g * 4 + 3] = __builtin_amdgcn_wmma_f32_16x16x32_f16(
          false, w3.v, false, bcur.v, (short)0, acc[g * 4 + 3], false, false);
    }
    wait_async_all();       // async tile for kt+1 fully in LDS
    __syncthreads();
    bcur = bnxt;
    buf ^= 1;
  }

  // Epilogue: lane holds row = rowBase + (lane&15), out-cols
  // colg*128 + ct*16 + hi*8 + r -> one packed b128 store per tile.
  const int rown = lane & 15;
  const int hi8  = hi * 8;
  _Float16* yrow = Y + (size_t)(rowBase + rown) * WIDTH + colg * 128;
  #pragma unroll
  for (int ct = 0; ct < 8; ++ct) {
    const float* bp = bias + colg * 128 + ct * 16 + hi8;
    H8 o;
    #pragma unroll
    for (int r = 0; r < 8; ++r) {
      float v = acc[ct][r] + bp[r];
      if (do_relu) v = fmaxf(v, 0.0f);
      o.h[r] = (_Float16)v;
    }
    *(uint4*)(yrow + ct * 16 + hi8) = o.u;
  }
}

// --------------------------- latent concat ---------------------------------
__global__ __launch_bounds__(128) void latent_kernel(
    const _Float16* __restrict__ H, const _Float16* __restrict__ DE,
    _Float16* __restrict__ LAT, int n0) {
  const long i  = blockIdx.x;
  const long gi = (long)n0 + i;
  const int ray = (int)(gi >> 7);
  const _Float16* h  = H  + (size_t)i * WIDTH;
  const _Float16* de = DE + (size_t)ray * DEPAD;
  _Float16* out = LAT + (size_t)i * LATKP;
  for (int j = threadIdx.x; j < LATKP / 4; j += 128) {   // 4 halves per thread
    H4 v;
    if (j < 64)        v.u = *(const uint2*)(h + j * 4);
    else if (j < 154)  v.u = *(const uint2*)(de + (j - 64) * 4);   // 616 = 154*4
    else               { v.u.x = 0u; v.u.y = 0u; }
    *(uint2*)(out + j * 4) = v.u;
  }
}

// ------------------------- final head projections --------------------------
__device__ __forceinline__ float sigmoidf_(float x) { return 1.0f / (1.0f + expf(-x)); }

__global__ __launch_bounds__(256) void head_out_kernel(
    const _Float16* __restrict__ HC, const _Float16* __restrict__ HA,
    const float* __restrict__ Wc, const float* __restrict__ bc,
    const float* __restrict__ Wa, const float* __restrict__ ba,
    float* __restrict__ out, int n0) {
  const int i = blockIdx.x * 256 + threadIdx.x;
  const long gi = (long)n0 + i;
  const _Float16* hc = HC + (size_t)i * WIDTH;
  const _Float16* ha = HA + (size_t)i * WIDTH;
  float c0 = bc[0], c1 = bc[1], c2 = bc[2], av = ba[0];
  for (int k8 = 0; k8 < WIDTH / 8; ++k8) {
    H8 a, b;
    a.u = *(const uint4*)(hc + k8 * 8);
    b.u = *(const uint4*)(ha + k8 * 8);
    #pragma unroll
    for (int j = 0; j < 8; ++j) {
      const int k = k8 * 8 + j;
      const float x = (float)a.h[j];
      c0 += x * Wc[k * 3 + 0];
      c1 += x * Wc[k * 3 + 1];
      c2 += x * Wc[k * 3 + 2];
      av += (float)b.h[j] * Wa[k];
    }
  }
  out[gi] = sigmoidf_(av);                          // alpha: first N floats
  float* cout = out + NPOINTS;                      // color: next 3N floats
  cout[gi * 3 + 0] = sigmoidf_(c0);
  cout[gi * 3 + 1] = sigmoidf_(c1);
  cout[gi * 3 + 2] = sigmoidf_(c2);
}

// ------------------------------- launcher ----------------------------------
extern "C" void kernel_launch(void* const* d_in, const int* in_sizes, int n_in,
                              void* d_out, int out_size, void* d_ws, size_t ws_size,
                              hipStream_t stream) {
  (void)in_sizes; (void)n_in; (void)out_size;

  const float* origins     = (const float*)d_in[0];
  const float* directions  = (const float*)d_in[1];
  const float* lengths     = (const float*)d_in[2];
  const float* hash_tables = (const float*)d_in[3];
  const float* mlp_in_W    = (const float*)d_in[4];
  const float* mlp_in_b    = (const float*)d_in[5];
  const float* mlp_hid_W   = (const float*)d_in[6];
  const float* mlp_hid_b   = (const float*)d_in[7];
  const float* color_in_W  = (const float*)d_in[8];
  const float* color_in_b  = (const float*)d_in[9];
  const float* color_hid_W = (const float*)d_in[10];
  const float* color_hid_b = (const float*)d_in[11];
  const float* color_out_W = (const float*)d_in[12];
  const float* color_out_b = (const float*)d_in[13];
  const float* alpha_in_W  = (const float*)d_in[14];
  const float* alpha_in_b  = (const float*)d_in[15];
  const float* alpha_hid_W = (const float*)d_in[16];
  const float* alpha_hid_b = (const float*)d_in[17];
  const float* alpha_out_W = (const float*)d_in[18];
  const float* alpha_out_b = (const float*)d_in[19];

  char* base = (char*)d_ws;
  size_t cur = 0;
  auto take = [&](size_t bytes) -> char* {
    char* p = base + cur;
    cur = (cur + bytes + 255) & ~(size_t)255;
    return p;
  };

  const int KT_L0 = 13, KT_H = 8, KT_L = 20;   // K tiles per layer kind
  _Float16* W0     = (_Float16*)take((size_t)KT_L0 * 8192 * 2);
  _Float16* Wh[3]  = { (_Float16*)take((size_t)KT_H * 8192 * 2),
                       (_Float16*)take((size_t)KT_H * 8192 * 2),
                       (_Float16*)take((size_t)KT_H * 8192 * 2) };
  _Float16* Wci    = (_Float16*)take((size_t)KT_L * 8192 * 2);
  _Float16* Wch[2] = { (_Float16*)take((size_t)KT_H * 8192 * 2),
                       (_Float16*)take((size_t)KT_H * 8192 * 2) };
  _Float16* Wai    = (_Float16*)take((size_t)KT_L * 8192 * 2);
  _Float16* Wah[2] = { (_Float16*)take((size_t)KT_H * 8192 * 2),
                       (_Float16*)take((size_t)KT_H * 8192 * 2) };
  _Float16* DE     = (_Float16*)take((size_t)RAYS * DEPAD * 2);

  // Pick a chunk size that fits the workspace (power-of-two divisor of N).
  const size_t fixed = cur;
  int CH = 32768;
  while (CH > 2048) {
    const size_t need = fixed +
        (size_t)CH * (ENCKP + 3 * WIDTH + LATKP) * 2 + 6 * 256;
    if (need <= ws_size) break;
    CH >>= 1;
  }
  _Float16* ENC = (_Float16*)take((size_t)CH * ENCKP * 2);
  _Float16* H0  = (_Float16*)take((size_t)CH * WIDTH * 2);
  _Float16* H1  = (_Float16*)take((size_t)CH * WIDTH * 2);
  _Float16* H2  = (_Float16*)take((size_t)CH * WIDTH * 2);
  _Float16* LAT = (_Float16*)take((size_t)CH * LATKP * 2);

  // --- one-time per call: weight swizzles + direction embedding ---
  swizzle_w_kernel<<<KT_L0, 256, 0, stream>>>(mlp_in_W, 392, W0);
  for (int j = 0; j < 3; ++j)
    swizzle_w_kernel<<<KT_H, 256, 0, stream>>>(mlp_hid_W + (size_t)j * 65536, 256, Wh[j]);
  swizzle_w_kernel<<<KT_L, 256, 0, stream>>>(color_in_W, 616, Wci);
  for (int j = 0; j < 2; ++j)
    swizzle_w_kernel<<<KT_H, 256, 0, stream>>>(color_hid_W + (size_t)j * 65536, 256, Wch[j]);
  swizzle_w_kernel<<<KT_L, 256, 0, stream>>>(alpha_in_W, 616, Wai);
  for (int j = 0; j < 2; ++j)
    swizzle_w_kernel<<<KT_H, 256, 0, stream>>>(alpha_hid_W + (size_t)j * 65536, 256, Wah[j]);
  dirembed_kernel<<<RAYS / 256, 256, 0, stream>>>(directions, DE);

  // --- chunked pipeline over the 262144 sample points ---
  const int nchunks = NPOINTS / CH;
  for (int c = 0; c < nchunks; ++c) {
    const int n0 = c * CH;
    encode_kernel<<<CH / 256, 256, 0, stream>>>(origins, directions, lengths,
                                                hash_tables, ENC, n0);
    // trunk MLP
    gemm_wmma_256<<<CH / 64, 256, 0, stream>>>(ENC, ENCKP, KT_L0, W0,  mlp_in_b,        H0, 1);
    gemm_wmma_256<<<CH / 64, 256, 0, stream>>>(H0,  WIDTH, KT_H, Wh[0], mlp_hid_b + 0,   H1, 1);
    gemm_wmma_256<<<CH / 64, 256, 0, stream>>>(H1,  WIDTH, KT_H, Wh[1], mlp_hid_b + 256, H0, 1);
    gemm_wmma_256<<<CH / 64, 256, 0, stream>>>(H0,  WIDTH, KT_H, Wh[2], mlp_hid_b + 512, H1, 1);
    latent_kernel<<<CH, 128, 0, stream>>>(H1, DE, LAT, n0);
    // color head trunk -> H0
    gemm_wmma_256<<<CH / 64, 256, 0, stream>>>(LAT, LATKP, KT_L, Wci,   color_in_b,        H0, 1);
    gemm_wmma_256<<<CH / 64, 256, 0, stream>>>(H0,  WIDTH, KT_H, Wch[0], color_hid_b + 0,   H2, 1);
    gemm_wmma_256<<<CH / 64, 256, 0, stream>>>(H2,  WIDTH, KT_H, Wch[1], color_hid_b + 256, H0, 1);
    // alpha head trunk -> H1
    gemm_wmma_256<<<CH / 64, 256, 0, stream>>>(LAT, LATKP, KT_L, Wai,   alpha_in_b,        H1, 1);
    gemm_wmma_256<<<CH / 64, 256, 0, stream>>>(H1,  WIDTH, KT_H, Wah[0], alpha_hid_b + 0,   H2, 1);
    gemm_wmma_256<<<CH / 64, 256, 0, stream>>>(H2,  WIDTH, KT_H, Wah[1], alpha_hid_b + 256, H1, 1);
    head_out_kernel<<<CH / 256, 256, 0, stream>>>(H0, H1, color_out_W, color_out_b,
                                                  alpha_out_W, alpha_out_b,
                                                  (float*)d_out, n0);
  }
}
